// Multi_head_attention_88184268521821
// MI455X (gfx1250) — compile-verified, matrix operand order checked by reading, not run on hardware
//
#include <hip/hip_runtime.h>
#include <hip/hip_bf16.h>
#include <hip/hip_fp16.h>

typedef __attribute__((ext_vector_type(16))) _Float16 v16h;
typedef __attribute__((ext_vector_type(8)))  float    v8f;
typedef __attribute__((ext_vector_type(4)))  int      v4i;

using gv4i = v4i __attribute__((address_space(1)));  // global int4
using lv4i = v4i __attribute__((address_space(3)));  // LDS int4

#define WMMA_F16(a, b, c) \
  __builtin_amdgcn_wmma_f32_16x16x32_f16(false, (a), false, (b), (short)0, (c), false, false)

#ifndef __has_builtin
#define __has_builtin(x) 0
#endif

#if __has_builtin(__builtin_amdgcn_global_load_async_to_lds_b128)
#define HAVE_ASYNC_LDS 1
#else
#define HAVE_ASYNC_LDS 0
#endif

#if __has_builtin(__builtin_amdgcn_s_wait_asynccnt)
#define WAIT_ASYNC() __builtin_amdgcn_s_wait_asynccnt(0)
#else
#define WAIT_ASYNC() asm volatile("s_wait_asynccnt 0" ::: "memory")
#endif

static constexpr int Bb  = 2;
static constexpr int Ss  = 2048;
static constexpr int HID = 1024;
static constexpr int Hh  = 16;
static constexpr int DH  = 64;

// ---------------------------------------------------------------------------
// Fragment loaders per CDNA5 ISA 7.12.2 (wave32):
//  A 16x32 f16: lane L holds row M=L&15; lanes 0-15 carry K={0..7,16..23},
//  lanes 16-31 carry K={8..15,24..31}. B 32x16 gathered symmetrically
//  (lane = column N, contiguous reduction dim in memory). Each lane loads
//  two contiguous 16-byte chunks.
// ---------------------------------------------------------------------------
__device__ inline v16h frag_from_f16(const _Float16* base, int ld, int k0) {
  const int lane = threadIdx.x & 31;
  const int r    = lane & 15;
  const int kb   = (lane & 16) ? 8 : 0;
  const _Float16* p = base + (size_t)r * ld + k0 + kb;
  v16h a;
#pragma unroll
  for (int i = 0; i < 8; ++i) a[i] = p[i];
#pragma unroll
  for (int i = 0; i < 8; ++i) a[8 + i] = p[16 + i];
  return a;
}

__device__ inline v16h frag_from_f32(const float* base, int ld, int k0) {
  const int lane = threadIdx.x & 31;
  const int r    = lane & 15;
  const int kb   = (lane & 16) ? 8 : 0;
  const float* p = base + (size_t)r * ld + k0 + kb;
  v16h a;
#pragma unroll
  for (int i = 0; i < 8; ++i) a[i] = (_Float16)p[i];
#pragma unroll
  for (int i = 0; i < 8; ++i) a[8 + i] = (_Float16)p[16 + i];
  return a;
}

// Row reductions across the 16-lane column group of a C fragment.
__device__ inline float rmax16(float v) {
#pragma unroll
  for (int m = 1; m < 16; m <<= 1) v = fmaxf(v, __shfl_xor(v, m, 32));
  return v;
}
__device__ inline float rsum16(float v) {
#pragma unroll
  for (int m = 1; m < 16; m <<= 1) v += __shfl_xor(v, m, 32);
  return v;
}

// ---------------------------------------------------------------------------
// Kernel 0: Wo f32 -> f16
// ---------------------------------------------------------------------------
__global__ void cvt_f16_kernel(const float* __restrict__ src,
                               _Float16* __restrict__ dst, int n) {
  int i = blockIdx.x * blockDim.x + threadIdx.x;
  if (i < n) dst[i] = (_Float16)src[i];
}

// ---------------------------------------------------------------------------
// Kernel 1: Q/K/V projections. Input viewed as flat [B*H*S, 64] row-major
// (the raw reshape keeps the 64-dim contiguous). q_i/k_i stored f16 in the
// same [B*H*S, 64] layout; v stored transposed as vT[B*H][64][S] so P*V
// B-fragments gather contiguously later.
// ---------------------------------------------------------------------------
__global__ __launch_bounds__(256) void proj_kernel(
    const float* __restrict__ q, const float* __restrict__ k,
    const float* __restrict__ v,
    const float* __restrict__ Wq, const float* __restrict__ bq,
    const float* __restrict__ Wk, const float* __restrict__ bk,
    const float* __restrict__ Wv, const float* __restrict__ bv,
    _Float16* __restrict__ qh, _Float16* __restrict__ kh,
    _Float16* __restrict__ vT) {
  const int wave = threadIdx.x >> 5;
  const int lane = threadIdx.x & 31;
  const int n    = lane & 15;
  const int hi   = lane >> 4;
  const size_t g0 = ((size_t)blockIdx.x * 8 + wave) * 16;  // row-tile base

  v16h aq0 = frag_from_f32(q + g0 * DH, DH, 0);
  v16h aq1 = frag_from_f32(q + g0 * DH, DH, 32);
  v16h ak0 = frag_from_f32(k + g0 * DH, DH, 0);
  v16h ak1 = frag_from_f32(k + g0 * DH, DH, 32);
  v16h av0 = frag_from_f32(v + g0 * DH, DH, 0);
  v16h av1 = frag_from_f32(v + g0 * DH, DH, 32);

  // q projection: out = X @ Wq^T + bq ; B[k][n] = Wq[n][k] (row n, contiguous k)
#pragma unroll
  for (int j = 0; j < 4; ++j) {
    v8f c = {};
    c = WMMA_F16(aq0, frag_from_f32(Wq + (size_t)(16 * j) * DH, DH, 0), c);
    c = WMMA_F16(aq1, frag_from_f32(Wq + (size_t)(16 * j) * DH, DH, 32), c);
    const int col = 16 * j + n;
    const float bias = bq[col];
#pragma unroll
    for (int r = 0; r < 8; ++r)
      qh[(g0 + r + 8 * hi) * DH + col] = (_Float16)(c[r] + bias);
  }
#pragma unroll
  for (int j = 0; j < 4; ++j) {
    v8f c = {};
    c = WMMA_F16(ak0, frag_from_f32(Wk + (size_t)(16 * j) * DH, DH, 0), c);
    c = WMMA_F16(ak1, frag_from_f32(Wk + (size_t)(16 * j) * DH, DH, 32), c);
    const int col = 16 * j + n;
    const float bias = bk[col];
#pragma unroll
    for (int r = 0; r < 8; ++r)
      kh[(g0 + r + 8 * hi) * DH + col] = (_Float16)(c[r] + bias);
  }
  // v projection -> transposed store
  const size_t bh = g0 >> 11;           // b*H + h (tiles never straddle heads)
  const int    s0 = (int)(g0 & 2047);
#pragma unroll
  for (int j = 0; j < 4; ++j) {
    v8f c = {};
    c = WMMA_F16(av0, frag_from_f32(Wv + (size_t)(16 * j) * DH, DH, 0), c);
    c = WMMA_F16(av1, frag_from_f32(Wv + (size_t)(16 * j) * DH, DH, 32), c);
    const int col = 16 * j + n;
    const float bias = bv[col];
#pragma unroll
    for (int r = 0; r < 8; ++r)
      vT[(bh * DH + col) * (size_t)Ss + (s0 + r + 8 * hi)] =
          (_Float16)(c[r] + bias);
  }
}

// ---------------------------------------------------------------------------
// Async (or fallback) staging of one 32-key K tile [32][64] and V tile
// [64][32] (from transposed V) into LDS. 256 threads x 16B each per tile.
// ---------------------------------------------------------------------------
__device__ inline void stage_kv(const _Float16* __restrict__ K,
                                const _Float16* __restrict__ V, int kb0,
                                _Float16* ktb, _Float16* vtb, int tid) {
  const int krow = tid >> 3, kseg = (tid & 7) * 8;  // key row, 8-f16 segment
  const int vrow = tid >> 2, vseg = (tid & 3) * 8;  // dim row, 8-f16 segment
  const _Float16* gk = K + (size_t)(kb0 + krow) * DH + kseg;
  const _Float16* gv = V + (size_t)vrow * Ss + kb0 + vseg;
  _Float16* lk = ktb + krow * DH + kseg;
  _Float16* lv = vtb + vrow * 32 + vseg;
#if HAVE_ASYNC_LDS
  __builtin_amdgcn_global_load_async_to_lds_b128((gv4i*)gk, (lv4i*)lk, 0, 0);
  __builtin_amdgcn_global_load_async_to_lds_b128((gv4i*)gv, (lv4i*)lv, 0, 0);
#else
  *(float4*)lk = *(const float4*)gk;
  *(float4*)lv = *(const float4*)gv;
#endif
}

// ---------------------------------------------------------------------------
// Kernel 2: flash attention. One wave owns a 16-row Q tile; 8 waves/block
// share double-buffered LDS K/V tiles staged with async global->LDS copies
// (DMA of block i+1 overlaps WMMA work on block i). Per 32-key block:
// 4 WMMA (scores), mask+scale, online softmax via lane shuffles, P relayout
// through LDS, 4 WMMA (P*V).
// ---------------------------------------------------------------------------
__global__ __launch_bounds__(256) void flash_kernel(
    const _Float16* __restrict__ qh, const _Float16* __restrict__ kh,
    const _Float16* __restrict__ vT, const int* __restrict__ mask,
    _Float16* __restrict__ wv) {
  __shared__ __align__(16) _Float16 kt[2][32 * DH];   // [key][dim]
  __shared__ __align__(16) _Float16 vt[2][DH * 32];   // [dim][key]
  __shared__ __align__(16) _Float16 plds[8][16 * 32]; // per-wave P tile

  const int tid  = threadIdx.x;
  const int wave = tid >> 5;
  const int lane = tid & 31;
  const int n    = lane & 15;
  const int hi   = lane >> 4;
  const int bh   = blockIdx.x >> 4;                    // b*H + h
  const int q0   = (blockIdx.x & 15) * 128 + wave * 16;
  const int b    = bh >> 4;

  const _Float16* Q  = qh + ((size_t)bh * Ss + q0) * DH;
  const _Float16* K  = kh + (size_t)bh * Ss * DH;
  const _Float16* V  = vT + (size_t)bh * DH * Ss;
  const int*      Mk = mask + (size_t)b * Ss * Ss;

  const v16h qf0 = frag_from_f16(Q, DH, 0);
  const v16h qf1 = frag_from_f16(Q, DH, 32);

  float mrow[8], lrow[8];
  v8f zacc = {};
  v8f o[4] = {zacc, zacc, zacc, zacc};
#pragma unroll
  for (int r = 0; r < 8; ++r) { mrow[r] = -3.0e38f; lrow[r] = 0.f; }

  // Prologue: stage first K/V block.
  stage_kv(K, V, 0, kt[0], vt[0], tid);
  WAIT_ASYNC();
  __syncthreads();

  int buf = 0;
  for (int kb0 = 0; kb0 < Ss; kb0 += 32) {
    // Kick off DMA for the next block into the other buffer (overlaps WMMA).
    if (kb0 + 32 < Ss) stage_kv(K, V, kb0 + 32, kt[buf ^ 1], vt[buf ^ 1], tid);

    const _Float16* Kt = kt[buf];
    const _Float16* Vt = vt[buf];

    // ---- scores: 16x32 tile = two 16x16 C fragments (from LDS K tile) -----
    v8f s0 = {}, s1 = {};
    s0 = WMMA_F16(qf0, frag_from_f16(Kt, DH, 0), s0);
    s0 = WMMA_F16(qf1, frag_from_f16(Kt, DH, 32), s0);
    s1 = WMMA_F16(qf0, frag_from_f16(Kt + 16 * DH, DH, 0), s1);
    s1 = WMMA_F16(qf1, frag_from_f16(Kt + 16 * DH, DH, 32), s1);

    // ---- mask + scale (faithful: mask!=0 -> -10000) -----------------------
#pragma unroll
    for (int r = 0; r < 8; ++r) {
      const size_t qr = (size_t)(q0 + r + 8 * hi) * Ss;
      const int m0 = Mk[qr + kb0 + n];
      const int m1 = Mk[qr + kb0 + 16 + n];
      s0[r] = m0 ? -10000.f : s0[r] * 0.125f;
      s1[r] = m1 ? -10000.f : s1[r] * 0.125f;
    }

    // ---- online softmax update -------------------------------------------
    float cr[8];
#pragma unroll
    for (int r = 0; r < 8; ++r) {
      const float tmax = rmax16(fmaxf(s0[r], s1[r]));
      const float mnew = fmaxf(mrow[r], tmax);
      const float corr = __expf(mrow[r] - mnew);
      const float p0   = __expf(s0[r] - mnew);
      const float p1   = __expf(s1[r] - mnew);
      lrow[r] = lrow[r] * corr + rsum16(p0 + p1);
      mrow[r] = mnew;
      cr[r]   = corr;
      s0[r]   = p0;
      s1[r]   = p1;
    }
#pragma unroll
    for (int j = 0; j < 4; ++j)
#pragma unroll
      for (int r = 0; r < 8; ++r) o[j][r] *= cr[r];

    // ---- P: C layout -> A layout via per-wave LDS region ------------------
    _Float16* pt = plds[wave];
#pragma unroll
    for (int r = 0; r < 8; ++r) {
      pt[(r + 8 * hi) * 32 + n]      = (_Float16)s0[r];
      pt[(r + 8 * hi) * 32 + 16 + n] = (_Float16)s1[r];
    }
    __syncthreads();
    const v16h pf = frag_from_f16(pt, 32, 0);

    // ---- O += P @ V  (B frags from LDS V tile, contiguous key chunks) -----
#pragma unroll
    for (int j = 0; j < 4; ++j) {
      const v16h vf = frag_from_f16(Vt + 16 * j * 32, 32, 0);
      o[j] = WMMA_F16(pf, vf, o[j]);
    }

    // Fence: our async stage for the next block is complete; barrier also
    // guarantees every wave finished reading the current buffers.
    WAIT_ASYNC();
    __syncthreads();
    buf ^= 1;
  }

  // ---- normalize and store wv (flat [B*H*S, 64] f16) ----------------------
#pragma unroll
  for (int j = 0; j < 4; ++j) {
    const int col = 16 * j + n;
#pragma unroll
    for (int r = 0; r < 8; ++r) {
      const size_t row = (size_t)bh * Ss + q0 + r + 8 * hi;
      wv[row * DH + col] = (_Float16)(o[j][r] / lrow[r]);
    }
  }
}

// ---------------------------------------------------------------------------
// Kernel 3: output projection. Raw-reshape merge means wv flat is exactly the
// [B*S, 1024] A matrix. out = A @ Wo^T + bo, M=4096 N=1024 K=1024.
// ---------------------------------------------------------------------------
__global__ __launch_bounds__(256) void outproj_kernel(
    const _Float16* __restrict__ wv, const _Float16* __restrict__ WoH,
    const float* __restrict__ bo, float* __restrict__ out) {
  const int wave = threadIdx.x >> 5;
  const int lane = threadIdx.x & 31;
  const int n    = lane & 15;
  const int hi   = lane >> 4;
  const int w    = blockIdx.x * 8 + wave;  // 0..4095
  const size_t row0 = (size_t)(w >> 4) * 16;
  const int    n0   = (w & 15) * 64;

  const _Float16* A = wv + row0 * HID;

  v8f zacc = {};
  v8f acc[4] = {zacc, zacc, zacc, zacc};

  for (int k0 = 0; k0 < HID; k0 += 32) {
    const v16h a = frag_from_f16(A, HID, k0);
#pragma unroll
    for (int j = 0; j < 4; ++j) {
      const v16h bf = frag_from_f16(WoH + (size_t)(n0 + 16 * j) * HID, HID, k0);
      acc[j] = WMMA_F16(a, bf, acc[j]);
    }
    if (k0 + 32 < HID)
      __builtin_prefetch(A + (size_t)(lane & 15) * HID + k0 + 32, 0, 3);
  }
#pragma unroll
  for (int j = 0; j < 4; ++j) {
    const int col  = n0 + 16 * j + n;
    const float bb = bo[col];
#pragma unroll
    for (int r = 0; r < 8; ++r)
      out[(row0 + r + 8 * hi) * HID + col] = acc[j][r] + bb;
  }
}

// ---------------------------------------------------------------------------
extern "C" void kernel_launch(void* const* d_in, const int* in_sizes, int n_in,
                              void* d_out, int out_size, void* d_ws,
                              size_t ws_size, hipStream_t stream) {
  const float* q    = (const float*)d_in[0];
  const float* k    = (const float*)d_in[1];
  const float* v    = (const float*)d_in[2];
  const int*   mask = (const int*)d_in[3];
  const float* Wq   = (const float*)d_in[4];
  const float* bq   = (const float*)d_in[5];
  const float* Wk   = (const float*)d_in[6];
  const float* bk   = (const float*)d_in[7];
  const float* Wv   = (const float*)d_in[8];
  const float* bv   = (const float*)d_in[9];
  const float* Wo   = (const float*)d_in[10];
  const float* bo   = (const float*)d_in[11];

  // Workspace layout (bytes): q_i | k_i | vT | wv (8 MB f16 each) | Wo f16 (2 MB)
  const size_t HEADELEMS = (size_t)Bb * Hh * Ss * DH;  // 4,194,304
  char* ws = (char*)d_ws;
  _Float16* qh  = (_Float16*)(ws + 0 * HEADELEMS);
  _Float16* kh  = (_Float16*)(ws + 2 * HEADELEMS);
  _Float16* vT  = (_Float16*)(ws + 4 * HEADELEMS);
  _Float16* wvb = (_Float16*)(ws + 6 * HEADELEMS);
  _Float16* WoH = (_Float16*)(ws + 8 * HEADELEMS);

  const int wo_n = HID * HID;  // 1,048,576
  cvt_f16_kernel<<<(wo_n + 255) / 256, 256, 0, stream>>>(Wo, WoH, wo_n);

  // 4096 row-tiles of 16 rows / 8 waves per block -> 512 blocks
  proj_kernel<<<512, 256, 0, stream>>>(q, k, v, Wq, bq, Wk, bk, Wv, bv,
                                       qh, kh, vT);

  // 32 (b,h) pairs * 16 blocks (128 q-rows each) = 512 blocks
  flash_kernel<<<512, 256, 0, stream>>>(qh, kh, vT, mask, wvb);

  // 4096 output tiles (16x64) / 8 waves per block = 512 blocks
  outproj_kernel<<<512, 256, 0, stream>>>(wvb, WoH, bo, (float*)d_out);
}
